// GeometricRound_8186207666896
// MI455X (gfx1250) — matrix operand order repaired and checked by reading, not compiled, not run on hardware
//
#include <hip/hip_runtime.h>
#include <stdint.h>

// ---------- types ----------
typedef __attribute__((ext_vector_type(16))) __bf16        v16bf;
typedef __attribute__((ext_vector_type(8)))  float         v8f;
typedef __attribute__((ext_vector_type(4)))  unsigned int  u32x4;
typedef __attribute__((ext_vector_type(8)))  int           i32x8;
typedef __attribute__((ext_vector_type(4)))  int           i32x4;
typedef __attribute__((ext_vector_type(4)))  float         f32x4;

union ABfrag { v16bf v; u32x4 q[2]; };

// Clifford Cl(3,0) Cayley table grouped by output blade k:
// for each k, 8 pairs (i=m, j=KJ[k*8+m]) with sign KS[k*8+m].
__constant__ int KJ[64] = {
  0,1,2,3,4,5,6,7,
  1,0,4,5,2,3,7,6,
  2,4,0,6,1,7,3,5,
  3,5,6,0,7,1,2,4,
  4,2,1,7,0,6,5,3,
  5,3,7,1,6,0,4,2,
  6,7,3,2,5,4,0,1,
  7,6,5,4,3,2,1,0 };
__constant__ float KS[64] = {
   1, 1, 1, 1,-1,-1,-1,-1,
   1, 1,-1,-1, 1, 1,-1,-1,
   1, 1, 1,-1,-1, 1, 1, 1,
   1, 1, 1, 1,-1,-1,-1,-1,
   1, 1,-1, 1, 1,-1, 1, 1,
   1, 1,-1,-1, 1, 1,-1,-1,
   1, 1, 1,-1,-1, 1, 1, 1,
   1, 1,-1, 1, 1,-1, 1, 1 };

__device__ __forceinline__ unsigned short f2bf(float f) {
  union { float f; unsigned int u; } x; x.f = f;
  unsigned int r = x.u + 0x7FFFu + ((x.u >> 16) & 1u);   // round-nearest-even
  return (unsigned short)(r >> 16);
}

// sigmoid via v_exp_f32 + v_rcp_f32 (TRANS ops, co-execute with WMMA)
__device__ __forceinline__ float fast_sigmoid(float v) {
  return __builtin_amdgcn_rcpf(1.f + __expf(-v));
}

// ---------- prologue: bf16-transpose weights, sw table, gate ----------
__global__ void prep_kernel(const float* __restrict__ iw,
                            const float* __restrict__ wg,
                            const float* __restrict__ wu,
                            const float* __restrict__ wd,
                            const float* __restrict__ ggate,
                            unsigned short* __restrict__ wgT,   // [512][128] bf16
                            unsigned short* __restrict__ wuT,   // [512][128] bf16
                            unsigned short* __restrict__ wdT,   // [128][512] bf16
                            float* __restrict__ swtab,          // [8][8]
                            float* __restrict__ gatep) {
  int idx = blockIdx.x * blockDim.x + threadIdx.x;
  int stride = gridDim.x * blockDim.x;
  for (int e = idx; e < 128 * 512; e += stride) {
    int d = e >> 9, f = e & 511;            // w_gate/w_up are [128][512] row-major
    wgT[f * 128 + d] = f2bf(wg[e]);
    wuT[f * 128 + d] = f2bf(wu[e]);
    wdT[d * 512 + f] = f2bf(wd[f * 128 + d]);   // w_down is [512][128]
  }
  if (blockIdx.x == 0 && threadIdx.x < 64) {
    int t = threadIdx.x;                    // t = k*8 + m, i = m
    int i = t & 7, j = KJ[t];
    swtab[t] = KS[t] * fast_sigmoid(iw[i * 8 + j]);
  }
  if (blockIdx.x == 0 && threadIdx.x == 64)
    gatep[0] = fast_sigmoid(ggate[0]);
}

// ---------- fused geo-product + LN + SwiGLU MLP + residual ----------
// Block = 256 threads = 8 wave32, processes 4 tokens (32 rows of D=128).
__launch_bounds__(256)
__global__ void geo_mlp_kernel(const float* __restrict__ x,
                               const unsigned short* __restrict__ wgT,
                               const unsigned short* __restrict__ wuT,
                               const unsigned short* __restrict__ wdT,
                               const float* __restrict__ swtab,
                               const float* __restrict__ gatep,
                               const float* __restrict__ gamma,
                               const float* __restrict__ beta,
                               float* __restrict__ out) {
  __shared__ __align__(16) float          xs[32 * 128];   // 16 KB  x tile (f32)
  __shared__ __align__(16) unsigned short ns[32 * 128];   //  8 KB  normed (bf16)
  __shared__ __align__(16) unsigned short hs[32 * 512];   // 32 KB  silu(g)*u (bf16)

  const int tid  = threadIdx.x;
  const int wave = tid >> 5;
  const int lane = tid & 31;
  const int row0 = blockIdx.x * 32;        // flat row = ((b*T+t)*8 + head)

  // ---- Phase A: TDM async load of contiguous 16KB x-tile into LDS ----
  if (wave == 0) {
    uint64_t gaddr = (uint64_t)(uintptr_t)(x + (size_t)row0 * 128);
    uint32_t laddr = (uint32_t)(uintptr_t)(&xs[0]);       // flat[31:0] == LDS offset
    const uint32_t UNITS = (32u * 128u * 4u) / 8u;        // 2048 x 8-byte units
    u32x4 g0; i32x8 g1;
    i32x4 g2 = {0, 0, 0, 0}, g3 = {0, 0, 0, 0};
    i32x8 g4 = {0, 0, 0, 0, 0, 0, 0, 0};
    g0[0] = 1u;                                           // count=1 (valid), user D#
    g0[1] = laddr;                                        // lds_addr
    g0[2] = (uint32_t)gaddr;                              // global_addr[31:0]
    g0[3] = (uint32_t)((gaddr >> 32) & 0x1FFFFFFu) | (2u << 30); // addr[56:32]|type=2
    g1[0] = (int)(3u << 16);                              // data_size = 8B
    g1[1] = (int)((UNITS & 0xFFFFu) << 16);               // tensor_dim0 lo16
    g1[2] = (int)((UNITS >> 16) | (1u << 16));            // dim0 hi16 | tensor_dim1=1
    g1[3] = (int)(UNITS << 16);                           // dim1 hi16=0 | tile_dim0
    g1[4] = (int)1u;                                      // tile_dim1=1, tile_dim2=0
    g1[5] = (int)UNITS;                                   // tensor_dim0_stride lo32
    g1[6] = 0;                                            // stride hi | dim1_stride lo
    g1[7] = 0;
    __builtin_amdgcn_tensor_load_to_lds(g0, g1, g2, g3, g4, 0);
    __builtin_amdgcn_s_wait_tensorcnt(0);
  }
  __syncthreads();

  // ---- Phase B: geometric product + gate-mix + LayerNorm -> ns (bf16) ----
  {
    const int token = wave >> 1;            // 2 waves per token
    const int kb    = (wave & 1) * 4;       // this wave owns output blades kb..kb+3
    const float gate = gatep[0];
    const float og   = 1.f - gate;
    const int d0 = lane * 4;                // 4 consecutive d per lane
    float xv[8][4];
    #pragma unroll
    for (int h = 0; h < 8; ++h) {
      const f32x4 t = *(const f32x4*)&xs[(token * 8 + h) * 128 + d0];
      xv[h][0] = t.x; xv[h][1] = t.y; xv[h][2] = t.z; xv[h][3] = t.w;
    }
    #pragma unroll
    for (int kk = 0; kk < 4; ++kk) {
      const int k = kb + kk;
      float m4[4] = {0.f, 0.f, 0.f, 0.f};
      #pragma unroll
      for (int m = 0; m < 8; ++m) {
        const float s = swtab[k * 8 + m];   // uniform -> scalar load
        const int j = KJ[k * 8 + m];
        #pragma unroll
        for (int c = 0; c < 4; ++c) m4[c] = fmaf(s * xv[m][c], xv[j][c], m4[c]);
      }
      float s1 = 0.f, s2 = 0.f;
      #pragma unroll
      for (int c = 0; c < 4; ++c) {
        float v = gate * m4[c] + og * xv[k][c];
        m4[c] = v; s1 += v; s2 += v * v;
      }
      #pragma unroll
      for (int off = 16; off >= 1; off >>= 1) {   // wave32 reduction
        s1 += __shfl_xor(s1, off, 32);
        s2 += __shfl_xor(s2, off, 32);
      }
      const float mu   = s1 * (1.f / 128.f);
      const float var  = s2 * (1.f / 128.f) - mu * mu;
      const float rstd = rsqrtf(var + 1e-5f);
      unsigned short nb[4];
      #pragma unroll
      for (int c = 0; c < 4; ++c)
        nb[c] = f2bf((m4[c] - mu) * rstd * gamma[d0 + c] + beta[d0 + c]);
      unsigned int* p = (unsigned int*)&ns[(token * 8 + k) * 128 + d0];
      p[0] = (unsigned int)nb[0] | ((unsigned int)nb[1] << 16);
      p[1] = (unsigned int)nb[2] | ((unsigned int)nb[3] << 16);
    }
  }
  __syncthreads();

  // ---- Phase C: GEMM1 (M=32,K=128,N=512), fused SwiGLU -> hs (bf16) ----
  {
    const int arow = (lane & 15);           // + mt*16 below
    const int akb  = (lane >> 4) << 3;      // A layout: K[0:8]/[16:24] vs K[8:16]/[24:32]
    const int bko  = (lane >> 4) << 3;      // B layout: K[0:8]/[16:24] vs K[8:16]/[24:32]
    #pragma unroll 1
    for (int q = 0; q < 8; ++q) {
      const int p  = wave + 8 * q;          // 64 (mt,nt) tile pairs
      const int mt = p >> 5, nt = p & 31;
      const int ar = mt * 16 + arow;
      const int bc = nt * 16 + (lane & 15);
      v8f accg = {0.f, 0.f, 0.f, 0.f, 0.f, 0.f, 0.f, 0.f};
      v8f accu = accg;
      #pragma unroll
      for (int kk = 0; kk < 4; ++kk) {
        ABfrag a, bg, bu;
        a.q[0] = *(const u32x4*)&ns[ar * 128 + kk * 32 + akb];
        a.q[1] = *(const u32x4*)&ns[ar * 128 + kk * 32 + akb + 16];
        const unsigned short* gp = wgT + bc * 128 + kk * 32 + bko;
        const unsigned short* up = wuT + bc * 128 + kk * 32 + bko;
        bg.q[0] = *(const u32x4*)gp; bg.q[1] = *(const u32x4*)(gp + 16);
        bu.q[0] = *(const u32x4*)up; bu.q[1] = *(const u32x4*)(up + 16);
        accg = __builtin_amdgcn_wmma_f32_16x16x32_bf16(false, a.v, false, bg.v,
                                                       (short)0, accg, false, false);
        accu = __builtin_amdgcn_wmma_f32_16x16x32_bf16(false, a.v, false, bu.v,
                                                       (short)0, accu, false, false);
      }
      const int crow0 = mt * 16 + ((lane >> 4) << 3);   // C layout rows
      const int ccol  = nt * 16 + (lane & 15);
      #pragma unroll
      for (int r = 0; r < 8; ++r) {
        const float g  = accg[r];
        const float hv = g * fast_sigmoid(g) * accu[r];   // silu(g)*u, rcp not div
        hs[(crow0 + r) * 512 + ccol] = f2bf(hv);
      }
    }
  }
  __syncthreads();

  // ---- Phase D: GEMM2 (M=32,K=512,N=128) + residual -> out ----
  {
    const int akb = (lane >> 4) << 3;
    const int bko = (lane >> 4) << 3;
    #pragma unroll 1
    for (int q = 0; q < 2; ++q) {
      const int p  = wave + 8 * q;          // 16 (mt,nt) tile pairs
      const int mt = p >> 3, nt = p & 7;
      const int ar = mt * 16 + (lane & 15);
      const int bc = nt * 16 + (lane & 15);
      v8f acc = {0.f, 0.f, 0.f, 0.f, 0.f, 0.f, 0.f, 0.f};
      #pragma unroll
      for (int kk = 0; kk < 16; ++kk) {
        ABfrag a, b;
        a.q[0] = *(const u32x4*)&hs[ar * 512 + kk * 32 + akb];
        a.q[1] = *(const u32x4*)&hs[ar * 512 + kk * 32 + akb + 16];
        const unsigned short* dp = wdT + bc * 512 + kk * 32 + bko;
        b.q[0] = *(const u32x4*)dp; b.q[1] = *(const u32x4*)(dp + 16);
        acc = __builtin_amdgcn_wmma_f32_16x16x32_bf16(false, a.v, false, b.v,
                                                      (short)0, acc, false, false);
      }
      const int crow0 = mt * 16 + ((lane >> 4) << 3);
      const int ccol  = nt * 16 + (lane & 15);
      #pragma unroll
      for (int r = 0; r < 8; ++r) {
        const int row = crow0 + r;
        out[(size_t)(row0 + row) * 128 + ccol] = xs[row * 128 + ccol] + acc[r];
      }
    }
  }
}

// ---------- launch ----------
extern "C" void kernel_launch(void* const* d_in, const int* in_sizes, int n_in,
                              void* d_out, int out_size, void* d_ws, size_t ws_size,
                              hipStream_t stream) {
  const float* x     = (const float*)d_in[0];
  const float* iw    = (const float*)d_in[1];
  const float* wg    = (const float*)d_in[2];
  const float* wu    = (const float*)d_in[3];
  const float* wd    = (const float*)d_in[4];
  const float* gamma = (const float*)d_in[5];
  const float* beta  = (const float*)d_in[6];
  const float* ggate = (const float*)d_in[7];

  unsigned short* wgT = (unsigned short*)d_ws;        // 128 KB
  unsigned short* wuT = wgT + 128 * 512;              // 128 KB
  unsigned short* wdT = wuT + 128 * 512;              // 128 KB
  float* swtab = (float*)(wdT + 128 * 512);           // 256 B
  float* gatep = swtab + 64;

  prep_kernel<<<256, 256, 0, stream>>>(iw, wg, wu, wd, ggate,
                                       wgT, wuT, wdT, swtab, gatep);

  const int rows   = in_sizes[0] / 128;               // B*T*N = 131072
  const int blocks = rows / 32;                       // 4096 blocks
  geo_mlp_kernel<<<blocks, 256, 0, stream>>>(x, wgT, wuT, wdT, swtab, gatep,
                                             gamma, beta, (float*)d_out);
}
